// MultiHeadAttention_77077483094398
// MI455X (gfx1250) — compile-verified, matrix operand order checked by reading, not run on hardware
//
#include <hip/hip_runtime.h>
#include <hip/hip_bf16.h>
#include <stdint.h>

// ---------------------------------------------------------------------------
// MHA layer for gfx1250 (MI455X): f16 WMMA (f32 accumulate) for all matmuls,
// fp32 softmax / LayerNorm. wave32. Data movement: TDM tensor_load_to_lds (Q),
// GLOBAL_LOAD_ASYNC_TO_LDS_B128 (K), double-buffered LDS staging.
// ---------------------------------------------------------------------------

typedef _Float16 half_t;
typedef __attribute__((ext_vector_type(16))) _Float16 v16h;
typedef __attribute__((ext_vector_type(8)))  float    v8f;
typedef __attribute__((ext_vector_type(4)))  uint32_t u32x4;
typedef __attribute__((ext_vector_type(8)))  int      i32x8;
typedef __attribute__((ext_vector_type(4)))  int      i32x4;

#define B_    4
#define S_    2048
#define D_    1024
#define H_    16
#define HD_   64
#define MT_   (B_ * S_)   // 8192 rows

union V16H { v16h v; uint32_t u[8]; _Float16 h[16]; };

// --- Load a 16x32 WMMA A/B operand from LDS, source stored row-major as
// element(r,k) = lds[r*stride + k].  (ISA 7.12.2 16-bit A layout.)  Works for
// both A (m,k) and B (n,k): B is consumed column-major == rows of (N,K).
__device__ inline v16h wmma_ld_rm(const half_t* lds, int stride) {
  const int lane = threadIdx.x & 31;
  const int hi   = lane >> 4;
  const int r    = lane & 15;
  V16H out;
  const half_t* row = lds + r * stride;
#pragma unroll
  for (int j = 0; j < 8; ++j) {
    int k = ((j & 3) << 1) + ((j >> 2) << 4) + (hi << 3);
    out.u[j] = *(const uint32_t*)(row + k);   // (k, k+1) pair, 4B-aligned
  }
  return out.v;
}

__device__ inline v8f wmma_f16(v16h a, v16h b, v8f c) {
  return __builtin_amdgcn_wmma_f32_16x16x32_f16(false, a, false, b,
                                                (short)0, c, false, false);
}

// --- CDNA5 async global->LDS copy (16B per lane), tracked with ASYNCcnt.
__device__ inline void async_ld_b128(uint32_t lds_byte_off, uint64_t gaddr) {
  asm volatile("global_load_async_to_lds_b128 %0, %1, off"
               :: "v"(lds_byte_off), "v"(gaddr)
               : "memory");
}
__device__ inline void wait_async0() {
  asm volatile("s_wait_asynccnt 0x0" ::: "memory");
}

// --- CDNA5 Tensor Data Mover: 2D f16 tile (tile_d0 x tile_d1 elems),
// global row stride `g_stride` elems, LDS row stride = tile_d0 + 8 halfs via
// D# padding (pad_interval=4 -> every 32 DWORDs, pad_amount=3 -> 4 DWORDs).
// Issue from ONE wave only (EXEC ignored; one issue moves the whole tile).
__device__ inline void tdm_load_tile_f16(uint32_t lds_byte, uint64_t g_byte,
                                         uint32_t tile_d0, uint32_t tile_d1,
                                         uint32_t g_stride) {
  u32x4 g0;
  g0[0] = 1u;                                    // count=1, user mode
  g0[1] = lds_byte;                              // lds_addr  (bits 63:32)
  g0[2] = (uint32_t)(g_byte & 0xffffffffu);      // global_addr[31:0]
  g0[3] = (uint32_t)((g_byte >> 32) & 0x01ffffffu)  // global_addr[56:32]
        | (2u << 30);                            // type=2 ("image")
  i32x8 g1;
  g1[0] = (1 << 16) | (1 << 20) | (4 << 22) | (3 << 25); // 2B elems, pad on
  g1[1] = (int)(tile_d0 << 16);                  // tensor_dim0[15:0] @63:48
  g1[2] = (int)(tile_d1 << 16);                  // dim0 hi=0; tensor_dim1[15:0]
  g1[3] = (int)(tile_d0 << 16);                  // dim1 hi=0; tile_dim0
  g1[4] = (int)(tile_d1);                        // tile_dim1; tile_dim2=0
  g1[5] = (int)(g_stride);                       // tensor_dim0_stride[31:0]
  g1[6] = 0;
  g1[7] = 0;
  i32x4 z4; z4[0] = 0; z4[1] = 0; z4[2] = 0; z4[3] = 0;
  i32x8 z8;
#pragma unroll
  for (int i = 0; i < 8; ++i) z8[i] = 0;
  // amdgpu-toolchain (clang-23) 6-arg form:
  // (u32x4 g0, i32x8 g1, i32x4 g2, i32x4 g3, i32x8, i32 cpol)
  __builtin_amdgcn_tensor_load_to_lds(g0, g1, z4, z4, z8, 0);
}

// ---------------------------------------------------------------------------
// GEMM:  Y[m][n] = sum_k X[m][k] * W[n][k] + bias[n]     (Y = X @ W^T + b)
// Block: 128 threads (4 waves), tile 128(M) x 64(N), K chunks of 32,
// register double-buffered global stream. Wave w: rows [32w,32w+32) x 64 N.
// ---------------------------------------------------------------------------
template <typename TIN, typename TOUT>
__global__ __launch_bounds__(128)
void gemm_xwt_kernel(const TIN* __restrict__ X, const float* __restrict__ W,
                     const float* __restrict__ bias, TOUT* __restrict__ Y,
                     int M, int N, int K) {
  constexpr int TM = 128, TN = 64, TK = 32, LS = TK + 2;  // 34-half stride
  __shared__ half_t As[TM][LS];
  __shared__ half_t Bs[TN][LS];

  const int tid  = threadIdx.x;
  const int wave = tid >> 5;
  const int lane = tid & 31;
  const int hi   = lane >> 4;
  const int nn   = lane & 15;
  const int m0   = blockIdx.x * TM;
  const int n0   = blockIdx.y * TN;
  const int br   = tid >> 1;                 // B stage row
  const int bc   = (tid & 1) << 4;           // B stage col 0/16

  half_t xa[32], wb[16];
  auto load_stage = [&](int k0) {
    const size_t xoff = (size_t)(m0 + tid) * K + k0;
#pragma unroll
    for (int i = 0; i < 32; ++i) xa[i] = (half_t)X[xoff + i];
    const size_t woff = (size_t)(n0 + br) * K + k0 + bc;
#pragma unroll
    for (int i = 0; i < 16; ++i) wb[i] = (half_t)W[woff + i];
  };

  v8f acc[2][4];
#pragma unroll
  for (int r = 0; r < 2; ++r)
#pragma unroll
    for (int t = 0; t < 4; ++t)
#pragma unroll
      for (int i = 0; i < 8; ++i) acc[r][t][i] = 0.0f;

  load_stage(0);
  for (int k0 = 0; k0 < K; k0 += TK) {
    __syncthreads();
#pragma unroll
    for (int i = 0; i < 32; ++i) As[tid][i] = xa[i];
#pragma unroll
    for (int i = 0; i < 16; ++i) Bs[br][bc + i] = wb[i];
    __syncthreads();

    if (k0 + TK < K) load_stage(k0 + TK);    // overlap next chunk with WMMAs

    const v16h a0 = wmma_ld_rm(&As[wave * 32][0],      LS);
    const v16h a1 = wmma_ld_rm(&As[wave * 32 + 16][0], LS);
#pragma unroll
    for (int t = 0; t < 4; ++t) {
      const v16h b = wmma_ld_rm(&Bs[t * 16][0], LS);
      acc[0][t] = wmma_f16(a0, b, acc[0][t]);
      acc[1][t] = wmma_f16(a1, b, acc[1][t]);
    }
  }

  // C/D layout: (vgpr i, lane<16) = row i ; (vgpr i, lane>=16) = row i+8
#pragma unroll
  for (int rr = 0; rr < 2; ++rr)
#pragma unroll
    for (int t = 0; t < 4; ++t) {
      const int n  = n0 + t * 16 + nn;
      const float bv = bias[n];
#pragma unroll
      for (int i = 0; i < 8; ++i) {
        const int m = m0 + wave * 32 + rr * 16 + i + hi * 8;
        Y[(size_t)m * N + n] = (TOUT)(acc[rr][t][i] + bv);
      }
    }
}

// ---------------------------------------------------------------------------
// Flash attention: one block = (b, h, 64-query tile). 128 threads / 4 waves.
// Q via TDM (padded stride in-hardware); K via async-LDS; V staged transposed.
// K and V^T are double-buffered (ping-pong) so chunk c+1's global traffic
// overlaps chunk c's WMMAs.
// Dynamic LDS (half units, stride 72 = 144B rows; 16B-aligned, 144r mod 256
// distinct for r=0..15 -> conflict-free 16-lane row reads):
//   Qs  @ 0      Ks[2] @ 4608/9216   VsT[2] @ 13824/18432   Ps @ 23040
// ---------------------------------------------------------------------------
#define LST   72
#define SM_Q  0
#define SM_K0 (64 * LST)
#define SM_K1 (2 * 64 * LST)
#define SM_V0 (3 * 64 * LST)
#define SM_V1 (4 * 64 * LST)
#define SM_P  (5 * 64 * LST)
#define SMEM_BYTES (6 * 64 * LST * 2)   // 55296 B

__global__ __launch_bounds__(128)
void attention_kernel(const half_t* __restrict__ Qh, const half_t* __restrict__ Kh,
                      const half_t* __restrict__ Vh, half_t* __restrict__ Ah) {
  extern __shared__ char smem[];
  half_t* sh = (half_t*)smem;

  const int tid  = threadIdx.x;
  const int wave = tid >> 5;
  const int lane = tid & 31;
  const int hi   = lane >> 4;
  const int nn   = lane & 15;
  const int q0   = blockIdx.x * 64;
  const int h    = blockIdx.y;
  const int b    = blockIdx.z;

  // --- stage Q block via Tensor Data Mover (one issue from wave 0) ---------
  if (wave == 0) {
    const uint64_t g = (uint64_t)(uintptr_t)(Qh + ((size_t)b * S_ + q0) * D_ + h * HD_);
    tdm_load_tile_f16((uint32_t)(SM_Q * 2), g, 64, 64, D_);
    __builtin_amdgcn_s_wait_tensorcnt(0);
  }
  __syncthreads();
  const v16h qa0 = wmma_ld_rm(sh + SM_Q + (wave * 16) * LST,      LST);
  const v16h qa1 = wmma_ld_rm(sh + SM_Q + (wave * 16) * LST + 32, LST);

  // --- stage helper for key/value chunk kc into buffer buf -----------------
  const int vr = tid >> 1;                 // V stage: key row
  const int vc = (tid & 1) << 5;           // V stage: d offset 0/32
  auto issue_stage = [&](int kc, int buf) {
    const uint32_t kbyte = (buf ? SM_K1 : SM_K0) * 2;
    const int      vbase = (buf ? SM_V1 : SM_V0);
    // K: 64 rows x 128B = 512 x 16B segments, 4/thread (async)
#pragma unroll
    for (int j = 0; j < 4; ++j) {
      const int s = tid + j * 128;
      const int r = s >> 3, seg = s & 7;
      const uint64_t g = (uint64_t)(uintptr_t)(Kh + ((size_t)b * S_ + kc + r) * D_ + h * HD_)
                       + (uint64_t)(seg * 16);
      async_ld_b128(kbyte + (uint32_t)(r * 144 + seg * 16), g);
    }
    // V transposed: VsT[d][k] = V[k][d]
    const size_t off = ((size_t)b * S_ + kc + vr) * D_ + h * HD_ + vc;
#pragma unroll
    for (int i = 0; i < 32; ++i)
      sh[vbase + (vc + i) * LST + vr] = Vh[off + i];
  };

  v8f o[4];
  float mrow[8], lrow[8];
#pragma unroll
  for (int t = 0; t < 4; ++t)
#pragma unroll
    for (int i = 0; i < 8; ++i) o[t][i] = 0.0f;
#pragma unroll
  for (int i = 0; i < 8; ++i) { mrow[i] = -1e30f; lrow[i] = 0.0f; }

  issue_stage(0, 0);
  for (int c = 0; c < S_ / 64; ++c) {
    const int buf = c & 1;
    wait_async0();
    __syncthreads();                      // chunk c visible everywhere
    if (c + 1 < S_ / 64) issue_stage((c + 1) * 64, buf ^ 1);  // overlap

    const half_t* ks = sh + (buf ? SM_K1 : SM_K0);
    const half_t* vt = sh + (buf ? SM_V1 : SM_V0);

    // scores S = Q K^T / sqrt(hd): 4 N-tiles x 2 k-chunks
    v8f s[4];
#pragma unroll
    for (int t = 0; t < 4; ++t) {
#pragma unroll
      for (int i = 0; i < 8; ++i) s[t][i] = 0.0f;
      s[t] = wmma_f16(qa0, wmma_ld_rm(ks + (t * 16) * LST,      LST), s[t]);
      s[t] = wmma_f16(qa1, wmma_ld_rm(ks + (t * 16) * LST + 32, LST), s[t]);
    }
    const float scale = 0.125f;           // 1/sqrt(64)
#pragma unroll
    for (int t = 0; t < 4; ++t)
#pragma unroll
      for (int i = 0; i < 8; ++i) s[t][i] *= scale;

    // online softmax: vgpr slot i of this lane == query row (i + 8*hi)
    float newm[8], resc[8], rsum[8];
#pragma unroll
    for (int i = 0; i < 8; ++i) {
      float v = fmaxf(fmaxf(s[0][i], s[1][i]), fmaxf(s[2][i], s[3][i]));
#pragma unroll
      for (int off = 1; off < 16; off <<= 1) v = fmaxf(v, __shfl_xor(v, off, 32));
      newm[i] = fmaxf(mrow[i], v);
      resc[i] = __expf(mrow[i] - newm[i]);
      rsum[i] = 0.0f;
    }
#pragma unroll
    for (int t = 0; t < 4; ++t)
#pragma unroll
      for (int i = 0; i < 8; ++i) {
        const float p = __expf(s[t][i] - newm[i]);
        s[t][i] = p;
        rsum[i] += p;
      }
#pragma unroll
    for (int i = 0; i < 8; ++i) {
#pragma unroll
      for (int off = 1; off < 16; off <<= 1) rsum[i] += __shfl_xor(rsum[i], off, 32);
      lrow[i] = lrow[i] * resc[i] + rsum[i];
      mrow[i] = newm[i];
    }
#pragma unroll
    for (int t = 0; t < 4; ++t)
#pragma unroll
      for (int i = 0; i < 8; ++i) o[t][i] *= resc[i];

    // materialize P (f16) into this wave's LDS tile
    half_t* pw = sh + SM_P + (wave * 16) * LST;
#pragma unroll
    for (int t = 0; t < 4; ++t)
#pragma unroll
      for (int i = 0; i < 8; ++i)
        pw[(i + hi * 8) * LST + t * 16 + nn] = (half_t)s[t][i];

    // O += P V : A = P (q,k) row-major; B = VsT (d,k) row-major
    const v16h pa0 = wmma_ld_rm(pw,      LST);
    const v16h pa1 = wmma_ld_rm(pw + 32, LST);
#pragma unroll
    for (int dt = 0; dt < 4; ++dt) {
      o[dt] = wmma_f16(pa0, wmma_ld_rm(vt + (dt * 16) * LST,      LST), o[dt]);
      o[dt] = wmma_f16(pa1, wmma_ld_rm(vt + (dt * 16) * LST + 32, LST), o[dt]);
    }
  }

  // epilogue: O / l, store f16 to (B,S,D) attention-output buffer
  float invl[8];
#pragma unroll
  for (int i = 0; i < 8; ++i) invl[i] = 1.0f / lrow[i];
#pragma unroll
  for (int t = 0; t < 4; ++t)
#pragma unroll
    for (int i = 0; i < 8; ++i) {
      const int r = wave * 16 + i + hi * 8;
      const int d = t * 16 + nn;
      Ah[((size_t)b * S_ + q0 + r) * D_ + h * HD_ + d] = (half_t)(o[t][i] * invl[i]);
    }
}

// ---------------------------------------------------------------------------
// Residual + LayerNorm: one block per row (1024 elems, 256 threads x 4).
// ---------------------------------------------------------------------------
__global__ __launch_bounds__(256)
void ln_kernel(const float* __restrict__ q, const float* __restrict__ y,
               const float* __restrict__ gamma, const float* __restrict__ beta,
               float* __restrict__ out) {
  __shared__ float redS[8], redQ[8];
  const int row  = blockIdx.x;
  const int tid  = threadIdx.x;
  const int wave = tid >> 5;
  const int lane = tid & 31;
  const float* qr = q + (size_t)row * D_;
  const float* yr = y + (size_t)row * D_;

  float x[4], sum = 0.0f, sq = 0.0f;
#pragma unroll
  for (int i = 0; i < 4; ++i) {
    const int d = tid + i * 256;
    x[i] = qr[d] + yr[d];
    sum += x[i];
    sq  += x[i] * x[i];
  }
#pragma unroll
  for (int off = 1; off < 32; off <<= 1) {
    sum += __shfl_xor(sum, off, 32);
    sq  += __shfl_xor(sq,  off, 32);
  }
  if (lane == 0) { redS[wave] = sum; redQ[wave] = sq; }
  __syncthreads();
  float ts = 0.0f, tq = 0.0f;
#pragma unroll
  for (int w = 0; w < 8; ++w) { ts += redS[w]; tq += redQ[w]; }
  const float mu   = ts * (1.0f / D_);
  const float var  = tq * (1.0f / D_) - mu * mu;
  const float rstd = rsqrtf(var + 1e-5f);
#pragma unroll
  for (int i = 0; i < 4; ++i) {
    const int d = tid + i * 256;
    out[(size_t)row * D_ + d] = (x[i] - mu) * rstd * gamma[d] + beta[d];
  }
}

// ---------------------------------------------------------------------------
extern "C" void kernel_launch(void* const* d_in, const int* in_sizes, int n_in,
                              void* d_out, int out_size, void* d_ws, size_t ws_size,
                              hipStream_t stream) {
  const float* query = (const float*)d_in[0];
  const float* key   = (const float*)d_in[1];
  const float* value = (const float*)d_in[2];
  const float* Wq = (const float*)d_in[3];  const float* bq = (const float*)d_in[4];
  const float* Wk = (const float*)d_in[5];  const float* bk = (const float*)d_in[6];
  const float* Wv = (const float*)d_in[7];  const float* bv = (const float*)d_in[8];
  const float* Wo = (const float*)d_in[9];  const float* bo = (const float*)d_in[10];
  const float* gamma = (const float*)d_in[11];
  const float* beta  = (const float*)d_in[12];
  float* out = (float*)d_out;

  // workspace: [Qh | Kh | Vh | Ah] f16 (16MB each); proj-out f32 aliases Qh/Kh
  const size_t NEL = (size_t)MT_ * D_;     // 8,388,608
  half_t* Qh = (half_t*)d_ws;
  half_t* Kh = Qh + NEL;
  half_t* Vh = Kh + NEL;
  half_t* Ah = Vh + NEL;
  float*  Pf = (float*)d_ws;               // alias: Qh/Kh dead after attention

  dim3 gG(MT_ / 128, D_ / 64);             // 64 x 16
  gemm_xwt_kernel<float, half_t><<<gG, 128, 0, stream>>>(query, Wq, bq, Qh, MT_, D_, D_);
  gemm_xwt_kernel<float, half_t><<<gG, 128, 0, stream>>>(key,   Wk, bk, Kh, MT_, D_, D_);
  gemm_xwt_kernel<float, half_t><<<gG, 128, 0, stream>>>(value, Wv, bv, Vh, MT_, D_, D_);

  dim3 gA(S_ / 64, H_, B_);                // 32 x 16 x 4
  attention_kernel<<<gA, 128, SMEM_BYTES, stream>>>(Qh, Kh, Vh, Ah);

  gemm_xwt_kernel<half_t, float><<<gG, 128, 0, stream>>>(Ah, Wo, bo, Pf, MT_, D_, D_);

  ln_kernel<<<MT_, 256, 0, stream>>>(query, Pf, gamma, beta, out);
}
